// MemEffCrossAttentionWeight_8976481649129
// MI455X (gfx1250) — compile-verified, matrix operand order checked by reading, not run on hardware
//
#include <hip/hip_runtime.h>
#include <hip/hip_bf16.h>

typedef __attribute__((ext_vector_type(16))) _Float16 v16h;
typedef __attribute__((ext_vector_type(8)))  _Float16 v8h;
typedef __attribute__((ext_vector_type(8)))  float    v8f;
typedef __attribute__((ext_vector_type(4)))  float    f32x4;

#define NBATCH 4
#define NQ     4096
#define NK     4096
#define DIM    256
#define INNER  64
#define BQ     16            // q-rows per workgroup (one WMMA M tile)
#define NEG_INF (-3.0e38f)

// ---------------------------------------------------------------------------
// Kernel 1: projection  out[row, j] = (sum_kk x[row,kk] * W[kk,j]) * scale
// stored as f16 for the WMMA score GEMM.  ~0.5 GFLOP each, trivial.
// ---------------------------------------------------------------------------
__global__ __launch_bounds__(256) void proj_f16_kernel(
    const float* __restrict__ x, const float* __restrict__ W,
    _Float16* __restrict__ out, float scale, int nrows)
{
    int gid = blockIdx.x * 256 + threadIdx.x;
    int row = gid >> 6;          // 64 outputs per row
    int j   = gid & 63;
    if (row >= nrows) return;
    const float* xr = x + (size_t)row * DIM;
    float acc = 0.f;
#pragma unroll 8
    for (int kk = 0; kk < DIM; ++kk)
        acc = fmaf(xr[kk], W[kk * INNER + j], acc);
    out[gid] = (_Float16)(acc * scale);
}

// pack two contiguous 8-half chunks into a 16-half WMMA fragment
__device__ inline v16h pack_frag(const _Float16* p)
{
    // p points at K=base (16B aligned); second chunk at K=base+16
    v8h lo = *(const v8h*)(p);
    v8h hi = *(const v8h*)(p + 16);
    v16h r;
#pragma unroll
    for (int i = 0; i < 8; ++i) { r[i] = lo[i]; r[i + 8] = hi[i]; }
    return r;
}

// Branchless sorted insert (t0 >= t1 >= t2 >= t3): 7 v_min/v_max ops,
// no control flow, values stay in VGPRs (no scratch demotion).
__device__ inline void ins_top4(float s, float& t0, float& t1, float& t2, float& t3)
{
    float m;
    m = fmaxf(t0, s); s = fminf(t0, s); t0 = m;
    m = fmaxf(t1, s); s = fminf(t1, s); t1 = m;
    m = fmaxf(t2, s); s = fminf(t2, s); t2 = m;
    t3 = fmaxf(t3, s);
}

// ---------------------------------------------------------------------------
// Kernel 2: per workgroup = 16 q-rows of one batch.
//   Phase 1: WMMA f16 score GEMM (16 x 4096), full strip kept in 256KB LDS
//   Phase 2: per-row top-4 -> threshold / max / exact denom (ties included)
//   Phase 3: masked softmax, streamed out with non-temporal b128 stores
// ---------------------------------------------------------------------------
__global__ __launch_bounds__(256) void attn_topk_softmax_kernel(
    const _Float16* __restrict__ qh, const _Float16* __restrict__ kh,
    float* __restrict__ out)
{
    extern __shared__ float smem[];
    float* scores  = smem;                       // [16][4096]  256 KB
    float* scratch = smem + BQ * NK;             // [16][16][4] top4 / partials
    float* rstat   = scratch + BQ * 16 * 4;      // [48]: max | thresh | invden

    const int tid  = threadIdx.x;
    const int lane = tid & 31;
    const int wave = tid >> 5;                   // 8 waves
    const int b    = blockIdx.x >> 8;            // 256 q-tiles per batch
    const int q0   = (blockIdx.x & 255) * BQ;

    const int hsel = lane >> 4;                  // which K half-group this lane holds
    const int lrow = lane & 15;                  // M (for A) / N (for B) index

    // ---- A fragments: 16 qh rows, loop-invariant, kept in VGPRs ----
    const _Float16* pa = qh + ((size_t)(b * NQ + q0 + lrow)) * INNER;
    const v16h a0 = pack_frag(pa + 8 * hsel);        // K 0..31
    const v16h a1 = pack_frag(pa + 32 + 8 * hsel);   // K 32..63

    // ---- Phase 1: score strip via WMMA, 32 N-tiles per wave ----
    // Constant trip count -> scalar (non-divergent) loop control.
#pragma unroll 2
    for (int i = 0; i < (NK / 16) / 8; ++i) {
        const int n0 = (wave + i * 8) * 16;
        const _Float16* pb = kh + ((size_t)(b * NK + n0 + lrow)) * INNER;
        const v16h b0 = pack_frag(pb + 8 * hsel);
        const v16h b1 = pack_frag(pb + 32 + 8 * hsel);
        v8f c = {};
        c = __builtin_amdgcn_wmma_f32_16x16x32_f16(false, a0, false, b0,
                                                   (short)0, c, false, false);
        c = __builtin_amdgcn_wmma_f32_16x16x32_f16(false, a1, false, b1,
                                                   (short)0, c, false, false);
        const int col     = n0 + lrow;           // N position
        const int rowBase = hsel * 8;            // C layout: M = j + 8*(lane>=16)
#pragma unroll
        for (int j = 0; j < 8; ++j)
            scores[(rowBase + j) * NK + col] = c[j];
    }
    __syncthreads();

    // ---- Phase 2: top-4 per row (16 threads per row, vectorized LDS reads) --
    const int row = tid >> 4;
    const int sub = tid & 15;
    float t0 = NEG_INF, t1 = NEG_INF, t2 = NEG_INF, t3 = NEG_INF;
    for (int i = 0; i < 64; ++i) {
        const int c4 = sub + 16 * i;             // lanes hit disjoint bank groups
        f32x4 v = *(const f32x4*)&scores[row * NK + c4 * 4];
        ins_top4(v.x, t0, t1, t2, t3);
        ins_top4(v.y, t0, t1, t2, t3);
        ins_top4(v.z, t0, t1, t2, t3);
        ins_top4(v.w, t0, t1, t2, t3);
    }
    float* my = scratch + (row * 16 + sub) * 4;
    my[0] = t0; my[1] = t1; my[2] = t2; my[3] = t3;
    __syncthreads();

    if (sub == 0) {                              // merge 16 partial top-4s
        float m0 = NEG_INF, m1 = NEG_INF, m2 = NEG_INF, m3 = NEG_INF;
        const float* src = scratch + row * 64;
#pragma unroll 8
        for (int i = 0; i < 64; ++i) ins_top4(src[i], m0, m1, m2, m3);
        rstat[row]      = m0;                    // row max
        rstat[16 + row] = m3;                    // 4th-largest = threshold
    }
    __syncthreads();

    // ---- exact denominator: sum exp(s-max) over s >= thresh (ties included).
    // Survivors are ~4/4096 per row: skip exp at wave granularity via ballot.
    {
        const float mv = rstat[row], th = rstat[16 + row];
        float part = 0.f;
        for (int i = 0; i < 64; ++i) {
            const int c4 = sub + 16 * i;
            f32x4 v = *(const f32x4*)&scores[row * NK + c4 * 4];
            const int hit = (v.x >= th) || (v.y >= th) || (v.z >= th) || (v.w >= th);
            if (__any(hit)) {
                part += (v.x >= th) ? __expf(v.x - mv) : 0.f;
                part += (v.y >= th) ? __expf(v.y - mv) : 0.f;
                part += (v.z >= th) ? __expf(v.z - mv) : 0.f;
                part += (v.w >= th) ? __expf(v.w - mv) : 0.f;
            }
        }
        scratch[row * 16 + sub] = part;
    }
    __syncthreads();
    if (sub == 0) {
        float d = 0.f;
#pragma unroll
        for (int i = 0; i < 16; ++i) d += scratch[row * 16 + i];
        rstat[32 + row] = 1.0f / d;
    }
    __syncthreads();

    // ---- Phase 3: masked softmax, coalesced non-temporal streaming stores ---
    for (int it = 0; it < 64; ++it) {
        const int idx4 = tid + 256 * it;         // consecutive lanes -> coalesced
        const int r    = idx4 >> 10;             // 1024 f32x4 per row
        const int c4   = idx4 & 1023;
        f32x4 v = *(const f32x4*)&scores[r * NK + c4 * 4];
        const float mv = rstat[r], th = rstat[16 + r], inv = rstat[32 + r];
        f32x4 o = {0.f, 0.f, 0.f, 0.f};
        const int hit = (v.x >= th) || (v.y >= th) || (v.z >= th) || (v.w >= th);
        if (__any(hit)) {                        // rare path (~12% of waves' iters)
            o.x = (v.x >= th) ? __expf(v.x - mv) * inv : 0.0f;
            o.y = (v.y >= th) ? __expf(v.y - mv) * inv : 0.0f;
            o.z = (v.z >= th) ? __expf(v.z - mv) * inv : 0.0f;
            o.w = (v.w >= th) ? __expf(v.w - mv) * inv : 0.0f;
        }
        f32x4* dst = (f32x4*)(out + ((size_t)(b * NQ + q0 + r)) * NK + c4 * 4);
        __builtin_nontemporal_store(o, dst);
    }
}

// ---------------------------------------------------------------------------
extern "C" void kernel_launch(void* const* d_in, const int* in_sizes, int n_in,
                              void* d_out, int out_size, void* d_ws, size_t ws_size,
                              hipStream_t stream)
{
    const float* q  = (const float*)d_in[0];
    const float* k  = (const float*)d_in[1];
    // d_in[2] = v (unused, matching reference)
    const float* Wq = (const float*)d_in[3];
    const float* Wk = (const float*)d_in[4];
    float* out = (float*)d_out;

    _Float16* qh = (_Float16*)d_ws;                          // 2 MB
    _Float16* kh = qh + (size_t)NBATCH * NQ * INNER;         // 2 MB

    const int nrows = NBATCH * NQ;                           // 16384
    const int nblk  = (nrows * INNER) / 256;                 // 4096 blocks
    proj_f16_kernel<<<nblk, 256, 0, stream>>>(q, Wq, qh, 0.125f, nrows); // 64^-0.5
    proj_f16_kernel<<<nblk, 256, 0, stream>>>(k, Wk, kh, 1.0f,   nrows);

    const size_t lds_bytes = (size_t)(BQ * NK + BQ * 16 * 4 + 48) * sizeof(float);
    (void)hipFuncSetAttribute((const void*)attn_topk_softmax_kernel,
                              hipFuncAttributeMaxDynamicSharedMemorySize,
                              (int)lds_bytes);               // ~260 KB < 320 KB WGP LDS
    attn_topk_softmax_kernel<<<NBATCH * (NQ / BQ), 256, lds_bytes, stream>>>(qh, kh, out);
}
